// PanODE_10170482556948
// MI455X (gfx1250) — matrix operand order repaired and checked by reading, not compiled, not run on hardware
//
#include <hip/hip_runtime.h>
#include <hip/hip_bf16.h>
#include <math.h>

typedef __attribute__((ext_vector_type(16))) _Float16 v16h;
typedef __attribute__((ext_vector_type(8)))  _Float16 v8h;
typedef __attribute__((ext_vector_type(8)))  float    v8f;

#define BATCH 256
#define DDIM  512
#define HDIM  2048
#define MNODE 64
#define NCOEF 32
#define NITER 12
#define ROWS  (BATCH * MNODE)   // 16384 GEMM rows (b,m)
#define BD    (BATCH * DDIM)    // 131072 (b,d) pairs

#define LDSPITCH 40             // f16 elements per LDS weight row (bank-conflict-free)

// ---------------------------------------------------------------------------
// Table setup: phi[n][m] = cos(n*pi*m/63), weighted DCT table, phi_s[n][p]
// ---------------------------------------------------------------------------
__global__ void pan_setup_tables(const float* __restrict__ t_span, int P,
                                 float* __restrict__ phi,    // [32][64]
                                 float* __restrict__ wphi,   // [32][64]
                                 float* __restrict__ phi_s,  // [32][16]
                                 float* __restrict__ scal)   // scal[0]=half
{
    const float PI = 3.14159265358979323846f;
    int j = threadIdx.x;
    if (j < MNODE) {
        float theta = PI * (float)j / (float)(MNODE - 1);
        float w = (j == 0 || j == MNODE - 1) ? 0.5f : 1.0f;
        for (int n = 0; n < NCOEF; ++n) {
            float p = cosf((float)n * theta);
            phi[n * MNODE + j] = p;
            float s = (2.0f / (float)(MNODE - 1)) * w * ((n == 0) ? 0.5f : 1.0f);
            wphi[n * MNODE + j] = s * p;
        }
    }
    if (j >= 64 && j < 80) {
        int p = j - 64;
        float t0 = t_span[0], t1 = t_span[P - 1];
        if (p < P) {
            float tau = -1.0f + 2.0f * (t_span[p] - t0) / (t1 - t0);
            tau = fminf(1.0f, fmaxf(-1.0f, tau));
            float th = acosf(tau);
            for (int n = 0; n < NCOEF; ++n)
                phi_s[n * 16 + p] = cosf((float)n * th);
        } else {
            for (int n = 0; n < NCOEF; ++n) phi_s[n * 16 + p] = 0.0f;
        }
    }
    if (j == 0) scal[0] = 0.5f * (t_span[P - 1] - t_span[0]);
}

// ---------------------------------------------------------------------------
// One-time weight transpose + f16 convert: W[R][C] f32 -> Wt[C][R] f16
// ---------------------------------------------------------------------------
__global__ __launch_bounds__(256)
void pan_w_to_f16t(const float* __restrict__ W, _Float16* __restrict__ Wt,
                   int R, int C)
{
    size_t i = (size_t)blockIdx.x * 256 + threadIdx.x;   // i = c*R + r, r fast
    if (i >= (size_t)R * C) return;
    int r = (int)(i % R);
    int c = (int)(i / R);
    Wt[i] = (_Float16)W[(size_t)r * C + c];
}

// ---------------------------------------------------------------------------
// Init Chebyshev coefficients: B[.,.,0] = y0, rest zero
// ---------------------------------------------------------------------------
__global__ __launch_bounds__(256)
void pan_init_B(const float* __restrict__ y0, float* __restrict__ Bc)
{
    int idx = blockIdx.x * 256 + threadIdx.x;            // b*512+d
    float4* row = (float4*)(Bc + (size_t)idx * NCOEF);
    float4 z = make_float4(0.f, 0.f, 0.f, 0.f);
#pragma unroll
    for (int q = 0; q < 8; ++q) row[q] = z;
    Bc[(size_t)idx * NCOEF] = y0[idx];
}

// ---------------------------------------------------------------------------
// Evaluate at Lobatto nodes + convert to f16:
//   Y16[(b*64+m)*512 + d] = sum_n B[b,d,n] * phi[n][m]
// ---------------------------------------------------------------------------
__global__ __launch_bounds__(256)
void pan_eval_nodes(const float* __restrict__ Bc, const float* __restrict__ phi,
                    _Float16* __restrict__ Y16)
{
    __shared__ float sphi[NCOEF * MNODE];
    for (int i = threadIdx.x; i < NCOEF * MNODE; i += 256) sphi[i] = phi[i];
    __syncthreads();

    int idx = blockIdx.x * 256 + threadIdx.x;            // b*512+d
    int b = idx >> 9, d = idx & 511;

    float Br[NCOEF];
    const float4* bp = (const float4*)(Bc + (size_t)idx * NCOEF);
#pragma unroll
    for (int q = 0; q < 8; ++q) {
        float4 t = bp[q];
        Br[4 * q + 0] = t.x; Br[4 * q + 1] = t.y;
        Br[4 * q + 2] = t.z; Br[4 * q + 3] = t.w;
    }
    for (int m = 0; m < MNODE; ++m) {
        float acc = 0.f;
#pragma unroll
        for (int n = 0; n < NCOEF; ++n) acc += Br[n] * sphi[n * MNODE + m];
        Y16[((size_t)b * MNODE + m) * DDIM + d] = (_Float16)acc;
    }
}

// ---------------------------------------------------------------------------
// CDNA5 async copy helpers
// ---------------------------------------------------------------------------
__device__ __forceinline__ void pan_async_cp16(unsigned lds_off, const void* gptr)
{
    unsigned long long ga = (unsigned long long)(uintptr_t)gptr;
    asm volatile("global_load_async_to_lds_b128 %0, %1, off"
                 :: "v"(lds_off), "v"(ga) : "memory");
}

__device__ __forceinline__ v16h pan_combine8(v8h x, v8h y)
{
    v16h r;
#pragma unroll
    for (int i = 0; i < 8; ++i) { r[i] = x[i]; r[i + 8] = y[i]; }
    return r;
}

// ---------------------------------------------------------------------------
// WMMA GEMM: C[MTOT][NTOT] = act( A[MTOT][KTOT] @ Wt[NTOT][KTOT]^T + bias )
// A, Wt are f16 row-major; acc f32 via v_wmma_f32_16x16x32_f16.
// 128 threads = 4 waves; block tile 64 rows x 64 cols. Weight K-tiles are
// staged into LDS with global_load_async_to_lds_b128 in a 4-buffer pipeline
// (2 tiles in flight, ASYNCcnt-tracked, one barrier per K-step), consumed by
// all 4 waves via ds_load_b128. Buffer indices are compile-time (K-loop
// unrolled x4) so accumulators stay pinned in registers.
// ---------------------------------------------------------------------------
template <int KTOT, int NTOT, bool TANH, bool OUT16>
__global__ __launch_bounds__(128)
void pan_gemm_wmma(const _Float16* __restrict__ A,
                   const _Float16* __restrict__ Wt,
                   const float* __restrict__ bias,
                   void* __restrict__ Cout)
{
    // [buf][col_local * LDSPITCH + k_local], pitch 40 halves = 80B (16B-aligned,
    // stride 20 words mod 64 banks tiles the bank space -> conflict-free b128s)
    __shared__ __align__(16) _Float16 sB[4][64 * LDSPITCH];

    const int tid  = threadIdx.x;
    const int wave = tid >> 5;
    const int lane = tid & 31;
    const int half = lane >> 4;          // K-half selector
    const int l16  = lane & 15;
    const int colBase = blockIdx.x * 64;
    const int rowBase = blockIdx.y * 64 + wave * 16;

    // Staging: 64 cols x 32 k x 2B = 4KB per tile = 256 16B-chunks; each of
    // 128 threads owns chunks tid and tid+128. chunk c -> col c/4, koff (c%4)*8
    const int cA = tid, cB = tid + 128;
    const int colA = cA >> 2, koffA = (cA & 3) * 8;
    const int colB = cB >> 2, koffB = (cB & 3) * 8;
    const _Float16* gA = Wt + (size_t)(colBase + colA) * KTOT + koffA;
    const _Float16* gB = Wt + (size_t)(colBase + colB) * KTOT + koffB;
    unsigned ldsA[4], ldsB[4];
#pragma unroll
    for (int s = 0; s < 4; ++s) {
        ldsA[s] = (unsigned)(uintptr_t)&sB[s][colA * LDSPITCH + koffA];
        ldsB[s] = (unsigned)(uintptr_t)&sB[s][colB * LDSPITCH + koffB];
    }

    const int NK = KTOT / 32;            // 16 or 64 (multiple of 4)

    // Prologue: stage K-tiles 0 and 1 into buffers 0 and 1
    pan_async_cp16(ldsA[0], gA);
    pan_async_cp16(ldsB[0], gB);
    pan_async_cp16(ldsA[1], gA + 32);
    pan_async_cp16(ldsB[1], gB + 32);

    // A-fragment register pipeline (16x32 f16): lanes 0-15 hold K {0..7,16..23}
    // of the tile, lanes 16-31 hold K {8..15,24..31}
    const _Float16* Arow = A + (size_t)(rowBase + l16) * KTOT;
    v8h alo = *(const v8h*)(Arow + half * 8);
    v8h ahi = *(const v8h*)(Arow + 16 + half * 8);

    v8f acc[4] = {v8f{}, v8f{}, v8f{}, v8f{}};

    for (int ks = 0; ks < NK; ks += 4) {
#pragma unroll
        for (int u = 0; u < 4; ++u) {            // tile i uses buffer u = i & 3
            const int i  = ks + u;
            const int k0 = i * 32;
            if (i + 2 < NK) {
                // issue tile i+2 into buffer (u+2)&3; then wait so that only
                // tiles i+1, i+2 (4 copies) remain in flight -> tile i landed
                pan_async_cp16(ldsA[(u + 2) & 3], gA + k0 + 64);
                pan_async_cp16(ldsB[(u + 2) & 3], gB + k0 + 64);
                asm volatile("s_wait_asynccnt 0x4" ::: "memory");
            } else if (i + 1 < NK) {
                asm volatile("s_wait_asynccnt 0x2" ::: "memory");
            } else {
                asm volatile("s_wait_asynccnt 0x0" ::: "memory");
            }
            __syncthreads();                     // tile i visible to all waves
                                                 // (also fences buffer reuse:
                                                 // last reads of this buffer
                                                 // were 4 steps ago, 2 barriers back)
            v16h a = pan_combine8(alo, ahi);
            if (i + 1 < NK) {                    // prefetch next A fragment
                alo = *(const v8h*)(Arow + k0 + 32 + half * 8);
                ahi = *(const v8h*)(Arow + k0 + 48 + half * 8);
            }
#pragma unroll
            for (int j = 0; j < 4; ++j) {
                const _Float16* bp = &sB[u][(j * 16 + l16) * LDSPITCH + half * 8];
                v8h blo = *(const v8h*)bp;
                v8h bhi = *(const v8h*)(bp + 16);
                v16h bfrag = pan_combine8(blo, bhi);
                acc[j] = __builtin_amdgcn_wmma_f32_16x16x32_f16(
                    false, a, false, bfrag, (short)0, acc[j], false, false);
            }
        }
    }

    // Epilogue: D layout — VGPR r: lanes 0-15 -> M=r, lanes 16-31 -> M=r+8
#pragma unroll
    for (int j = 0; j < 4; ++j) {
        int col = colBase + j * 16 + l16;
        float bv = bias[col];
#pragma unroll
        for (int r = 0; r < 8; ++r) {
            int mrow = rowBase + r + half * 8;
            float v = acc[j][r] + bv;
            if (TANH) v = tanhf(v);
            if (OUT16)
                ((_Float16*)Cout)[(size_t)mrow * NTOT + col] = (_Float16)v;
            else
                ((float*)Cout)[(size_t)mrow * NTOT + col] = v;
        }
    }
}

// ---------------------------------------------------------------------------
// Weighted DCT of f at nodes + spectral antiderivative + boundary pin:
//   c[n]     = sum_m F[b,m,d] * wphi[n][m]            (scaling baked in)
//   br[k]    = half*(c[k-1]-c[k+1])/(2k), c[32]=0
//   br[0]    = y0 - sum_k br[k]*(-1)^k
// ---------------------------------------------------------------------------
__global__ __launch_bounds__(256)
void pan_dct_update(const float* __restrict__ F, const float* __restrict__ wphi,
                    const float* __restrict__ y0, const float* __restrict__ scal,
                    float* __restrict__ Bc)
{
    __shared__ float sw[NCOEF * MNODE];
    for (int i = threadIdx.x; i < NCOEF * MNODE; i += 256) sw[i] = wphi[i];
    __syncthreads();

    int idx = blockIdx.x * 256 + threadIdx.x;            // b*512+d
    int b = idx >> 9, d = idx & 511;

    float c[NCOEF];
#pragma unroll
    for (int n = 0; n < NCOEF; ++n) c[n] = 0.f;
    for (int m = 0; m < MNODE; ++m) {
        float f = F[((size_t)b * MNODE + m) * DDIM + d];
#pragma unroll
        for (int n = 0; n < NCOEF; ++n) c[n] += f * sw[n * MNODE + m];
    }

    float hf = scal[0];
    float br[NCOEF];
    float s = 0.f;
#pragma unroll
    for (int k = 1; k < NCOEF; ++k) {
        float ckp1 = (k + 1 < NCOEF) ? c[k + 1] : 0.f;
        float v = hf * (c[k - 1] - ckp1) / (2.0f * (float)k);
        br[k] = v;
        s += (k & 1) ? -v : v;
    }
    br[0] = y0[idx] - s;

    float4* rowp = (float4*)(Bc + (size_t)idx * NCOEF);
#pragma unroll
    for (int q = 0; q < 8; ++q)
        rowp[q] = make_float4(br[4 * q], br[4 * q + 1], br[4 * q + 2], br[4 * q + 3]);
}

// ---------------------------------------------------------------------------
// Final trajectory evaluation: out[p][b][d] = sum_n B[b,d,n] * phi_s[n][p]
// ---------------------------------------------------------------------------
__global__ __launch_bounds__(256)
void pan_traj_eval(const float* __restrict__ Bc, const float* __restrict__ phi_s,
                   float* __restrict__ out, int P)
{
    __shared__ float sp[NCOEF * 16];
    for (int i = threadIdx.x; i < NCOEF * 16; i += 256) sp[i] = phi_s[i];
    __syncthreads();

    int idx = blockIdx.x * 256 + threadIdx.x;            // b*512+d
    float Br[NCOEF];
    const float4* bp = (const float4*)(Bc + (size_t)idx * NCOEF);
#pragma unroll
    for (int q = 0; q < 8; ++q) {
        float4 t = bp[q];
        Br[4 * q + 0] = t.x; Br[4 * q + 1] = t.y;
        Br[4 * q + 2] = t.z; Br[4 * q + 3] = t.w;
    }
    for (int p = 0; p < P; ++p) {
        float acc = 0.f;
#pragma unroll
        for (int n = 0; n < NCOEF; ++n) acc += Br[n] * sp[n * 16 + p];
        out[(size_t)p * BD + idx] = acc;
    }
}

// ---------------------------------------------------------------------------
extern "C" void kernel_launch(void* const* d_in, const int* in_sizes, int n_in,
                              void* d_out, int out_size, void* d_ws, size_t ws_size,
                              hipStream_t stream)
{
    const float* y0     = (const float*)d_in[0];
    const float* t_span = (const float*)d_in[1];
    const float* W1     = (const float*)d_in[2];
    const float* b1     = (const float*)d_in[3];
    const float* W2     = (const float*)d_in[4];
    const float* b2     = (const float*)d_in[5];
    float* out = (float*)d_out;
    const int P = in_sizes[1];

    char* ws = (char*)d_ws;
    size_t off = 0;
    auto alloc = [&](size_t bytes) -> void* {
        void* p = ws + off;
        off += (bytes + 255) & ~(size_t)255;
        return p;
    };
    float*    phi   = (float*)alloc(NCOEF * MNODE * 4);
    float*    wphi  = (float*)alloc(NCOEF * MNODE * 4);
    float*    phi_s = (float*)alloc(NCOEF * 16 * 4);
    float*    scal  = (float*)alloc(256);
    _Float16* W1t   = (_Float16*)alloc((size_t)HDIM * DDIM * 2); // [2048][512]
    _Float16* W2t   = (_Float16*)alloc((size_t)DDIM * HDIM * 2); // [512][2048]
    float*    Bc    = (float*)alloc((size_t)BD * NCOEF * 4);     // [b*d][32]
    _Float16* Y16   = (_Float16*)alloc((size_t)ROWS * DDIM * 2); // [(b,m)][512]
    _Float16* H16   = (_Float16*)alloc((size_t)ROWS * HDIM * 2); // [(b,m)][2048]
    float*    F     = (float*)alloc((size_t)ROWS * DDIM * 4);    // [(b,m)][512]

    pan_setup_tables<<<1, 128, 0, stream>>>(t_span, P, phi, wphi, phi_s, scal);
    pan_w_to_f16t<<<(DDIM * HDIM + 255) / 256, 256, 0, stream>>>(W1, W1t, DDIM, HDIM);
    pan_w_to_f16t<<<(HDIM * DDIM + 255) / 256, 256, 0, stream>>>(W2, W2t, HDIM, DDIM);
    pan_init_B<<<BD / 256, 256, 0, stream>>>(y0, Bc);

    for (int it = 0; it < NITER; ++it) {
        pan_eval_nodes<<<BD / 256, 256, 0, stream>>>(Bc, phi, Y16);
        // H = tanh(Y @ W1 + b1): M=16384, K=512, N=2048 -> f16 out
        pan_gemm_wmma<DDIM, HDIM, true, true>
            <<<dim3(HDIM / 64, ROWS / 64), 128, 0, stream>>>(Y16, W1t, b1, H16);
        // F = H @ W2 + b2: M=16384, K=2048, N=512 -> f32 out
        pan_gemm_wmma<HDIM, DDIM, false, false>
            <<<dim3(DDIM / 64, ROWS / 64), 128, 0, stream>>>(H16, W2t, b2, F);
        pan_dct_update<<<BD / 256, 256, 0, stream>>>(F, wphi, y0, scal, Bc);
    }

    pan_traj_eval<<<BD / 256, 256, 0, stream>>>(Bc, phi_s, out, P);
}
